// GraphNet_13443247636814
// MI455X (gfx1250) — matrix operand order repaired and checked by reading, not compiled, not run on hardware
//
#include <hip/hip_runtime.h>
#include <stdint.h>

#define N_NODES 50000
#define N_EDGES 350000
#define LN_EPS 1e-5f

typedef __bf16 bf16;
typedef __attribute__((ext_vector_type(8)))  __bf16 v8bf;
typedef __attribute__((ext_vector_type(16))) __bf16 v16bf;
typedef __attribute__((ext_vector_type(8)))  float  v8f;

__device__ __forceinline__ bf16 f2bf(float f) {
  unsigned u = __builtin_bit_cast(unsigned, f);
  unsigned r = u + 0x7FFFu + ((u >> 16) & 1u);
  unsigned short h = (unsigned short)(r >> 16);
  return __builtin_bit_cast(bf16, h);
}

// ---------------------------------------------------------------- utilities
__global__ void k_zero(float* __restrict__ p, long long n) {
  long long i = (long long)blockIdx.x * blockDim.x + threadIdx.x;
  if (i < n) p[i] = 0.f;
}
__global__ void k_copy(const float* __restrict__ a, float* __restrict__ b, long long n) {
  long long i = (long long)blockIdx.x * blockDim.x + threadIdx.x;
  if (i < n) b[i] = a[i];
}
__global__ void k_relu(float* __restrict__ p, long long n) {
  long long i = (long long)blockIdx.x * blockDim.x + threadIdx.x;
  if (i < n) p[i] = fmaxf(p[i], 0.f);
}
__global__ void k_f32_to_bf16(const float* __restrict__ in, bf16* __restrict__ out, long long n) {
  long long i = (long long)blockIdx.x * blockDim.x + threadIdx.x;
  if (i < n) out[i] = f2bf(in[i]);
}

// ------------------------------------------------------------- gcn_norm
__global__ void k_deg(float* __restrict__ deg, const long long* __restrict__ col,
                      const float* __restrict__ w, int E) {
  int e = blockIdx.x * blockDim.x + threadIdx.x;
  if (e < E) atomicAdd(&deg[col[e]], w[e]);
}
__global__ void k_norm(const float* __restrict__ deg, const long long* __restrict__ row,
                       const long long* __restrict__ col, const float* __restrict__ w,
                       float* __restrict__ nrm, int E) {
  int e = blockIdx.x * blockDim.x + threadIdx.x;
  if (e >= E) return;
  float dr = deg[row[e]], dc = deg[col[e]];
  float ir = dr > 0.f ? rsqrtf(dr) : 0.f;
  float ic = dc > 0.f ? rsqrtf(dc) : 0.f;
  nrm[e] = ir * w[e] * ic;
}

// ------------------------------------------------------------- scatter hop
__global__ void k_scatter4(const float* __restrict__ hin, float* __restrict__ hout,
                           const long long* __restrict__ row, const long long* __restrict__ col,
                           const float* __restrict__ nrm, int C, long long total) {
  long long idx = (long long)blockIdx.x * blockDim.x + threadIdx.x;
  if (idx >= total) return;
  int chunks = C >> 2;
  long long e = idx / chunks;
  int c = (int)(idx - e * chunks) * 4;
  float w = nrm[e];
  const float* s = hin + (size_t)row[e] * C + c;
  float* d = hout + (size_t)col[e] * C + c;
  atomicAdd(d + 0, w * s[0]);
  atomicAdd(d + 1, w * s[1]);
  atomicAdd(d + 2, w * s[2]);
  atomicAdd(d + 3, w * s[3]);
}
__global__ void k_scatter1(const float* __restrict__ hin, float* __restrict__ hout,
                           const long long* __restrict__ row, const long long* __restrict__ col,
                           const float* __restrict__ nrm, int E) {
  int e = blockIdx.x * blockDim.x + threadIdx.x;
  if (e < E) atomicAdd(&hout[col[e]], nrm[e] * hin[row[e]]);
}

// ------------------------------------------------------------- small scalar GEMM
__global__ void k_sgemm(const float* __restrict__ A, const float* __restrict__ W,
                        const float* __restrict__ bias, float* __restrict__ Out,
                        int M, int K, int Nc, int accumulate, int addbias) {
  long long idx = (long long)blockIdx.x * blockDim.x + threadIdx.x;
  if (idx >= (long long)M * Nc) return;
  int m = (int)(idx / Nc);
  int n = (int)(idx - (long long)m * Nc);
  float a = accumulate ? Out[idx] : (addbias ? bias[n] : 0.f);
  const float* ar = A + (size_t)m * K;
  for (int k = 0; k < K; k++) a += ar[k] * W[(size_t)k * Nc + n];
  Out[idx] = a;
}

// ------------------------------------------------------------- WMMA B-matrix pack
// Packs fp32 W[row_off + K, N] (row stride = ld) into the CDNA5 16-bit B-frag
// lane layout: lanes 0-15 hold K = ks*32+{0..7,16..23} (N = lane), lanes 16-31 hold
// K = ks*32+{8..15,24..31} (N = lane-16); 16 bf16 per lane.
__global__ void k_pack_w(const float* __restrict__ W, bf16* __restrict__ Bp,
                         int Kc, int Nc, int ld, int row_off, int total) {
  int idx = blockIdx.x * blockDim.x + threadIdx.x;
  if (idx >= total) return;
  int lane = idx & 31;
  int tile = idx >> 5;
  int ntiles = Nc >> 4;
  int nb = tile % ntiles;
  int ks = tile / ntiles;
  int kbase = ks * 32 + ((lane & 16) ? 8 : 0);
  int nv = nb * 16 + (lane & 15);
  bf16* dst = Bp + (size_t)idx * 16;
#pragma unroll
  for (int i = 0; i < 16; i++)
    dst[i] = f2bf(W[(size_t)(row_off + kbase + i + (i >= 8 ? 8 : 0)) * ld + nv]);
}

// ------------------------------------------------------------- WMMA GEMM
// One wave per 16x64 output tile (4 n-tiles share one A fragment -> 4x A reuse).
// A: bf16 row-major [M x 32*Ksteps], Bp packed per k_pack_w.
__global__ void k_wmma_gemm(const bf16* __restrict__ A, const bf16* __restrict__ Bp,
                            const float* __restrict__ bias, float* __restrict__ Out,
                            int Ksteps, int Nc, int accumulate, int addbias) {
  int lane = threadIdx.x;
  int tm = blockIdx.x;
  int tn0 = blockIdx.y * 4;
  int mloc = lane & 15;
  int base0 = (lane & 16) ? 8 : 0;
  int K = Ksteps * 32;
  const bf16* Arow = A + (size_t)(tm * 16 + mloc) * K;
  v8f c[4];
  if (accumulate) {
#pragma unroll
    for (int t = 0; t < 4; t++) {
      int n = (tn0 + t) * 16 + mloc;
#pragma unroll
      for (int v = 0; v < 8; v++) c[t][v] = Out[(size_t)(tm * 16 + base0 + v) * Nc + n];
    }
  } else if (addbias) {
#pragma unroll
    for (int t = 0; t < 4; t++) {
      float bv = bias[(tn0 + t) * 16 + mloc];
#pragma unroll
      for (int v = 0; v < 8; v++) c[t][v] = bv;
    }
  } else {
#pragma unroll
    for (int t = 0; t < 4; t++)
#pragma unroll
      for (int v = 0; v < 8; v++) c[t][v] = 0.f;
  }
  size_t bstride = (size_t)(Nc >> 4) * 512;  // elements per K-step of packed B
  const bf16* bp = Bp + ((size_t)tn0 * 32 + lane) * 16;
  for (int ks = 0; ks < Ksteps; ks++) {
    v8bf lo = *(const v8bf*)(Arow + ks * 32 + base0);
    v8bf hi = *(const v8bf*)(Arow + ks * 32 + base0 + 16);
    v16bf a;
#pragma unroll
    for (int i = 0; i < 8; i++) { a[i] = lo[i]; a[8 + i] = hi[i]; }
    const bf16* bk = bp + (size_t)ks * bstride;
#pragma unroll
    for (int t = 0; t < 4; t++) {
      v16bf b = *(const v16bf*)(bk + (size_t)t * 512);
      c[t] = __builtin_amdgcn_wmma_f32_16x16x32_bf16(false, a, false, b, (short)0,
                                                     c[t], false, false);
    }
  }
#pragma unroll
  for (int t = 0; t < 4; t++) {
    int n = (tn0 + t) * 16 + mloc;
#pragma unroll
    for (int v = 0; v < 8; v++) Out[(size_t)(tm * 16 + base0 + v) * Nc + n] = c[t][v];
  }
}

// ------------------------------------------------------------- small edge MLP (e1/e2)
template <int CIN, int HID>
__global__ void k_edge_small(const float* __restrict__ h, const long long* __restrict__ row,
                             const long long* __restrict__ col, const float* __restrict__ ea,
                             const float* __restrict__ W1, const float* __restrict__ b1,
                             const float* __restrict__ g, const float* __restrict__ be,
                             const float* __restrict__ W2, const float* __restrict__ b2,
                             float* __restrict__ out, int E) {
  int e = blockIdx.x * blockDim.x + threadIdx.x;
  if (e >= E) return;
  const float* s = h + (size_t)row[e] * CIN;
  const float* d = h + (size_t)col[e] * CIN;
  float a = ea[e];
  float hid[HID];
  float mu = 0.f;
#pragma unroll 1
  for (int j = 0; j < HID; j++) {
    float acc = b1[j] + a * W1[2 * CIN * HID + j];
    for (int i = 0; i < CIN; i++)
      acc += s[i] * W1[i * HID + j] + d[i] * W1[(CIN + i) * HID + j];
    acc = fmaxf(acc, 0.f);
    hid[j] = acc;
    mu += acc;
  }
  mu *= (1.f / HID);
  float var = 0.f;
  for (int j = 0; j < HID; j++) { float t = hid[j] - mu; var += t * t; }
  var *= (1.f / HID);
  float rs = rsqrtf(var + LN_EPS);
  float o = b2[0];
  for (int j = 0; j < HID; j++) o += ((hid[j] - mu) * rs * g[j] + be[j]) * W2[j];
  out[e] = o;
}

// ------------------------------------------------------------- fused e3 edge MLP (WMMA)
// 64 edges per block; 8 waves: wave w -> edge-subtile (w>>1), hid-half (w&1)
// covering 4 n-tiles. One gathered A fragment feeds 4 WMMAs (src + dst halves
// of W1). C=512, HID=128. LayerNorm + W2 reduction done in LDS.
__global__ void k_edge_wmma(const bf16* __restrict__ hb, const long long* __restrict__ row,
                            const long long* __restrict__ col, const float* __restrict__ ea,
                            const bf16* __restrict__ BpS, const bf16* __restrict__ BpD,
                            const float* __restrict__ W1last, const float* __restrict__ b1,
                            const float* __restrict__ g, const float* __restrict__ be,
                            const float* __restrict__ W2, const float* __restrict__ b2,
                            float* __restrict__ out) {
  __shared__ float hsm[64][132];
  int lane = threadIdx.x & 31;
  int wave = threadIdx.x >> 5;  // 0..7
  int mt = wave >> 1;           // edge sub-tile 0..3
  int nh = wave & 1;            // hid half -> n-tiles nh*4 .. nh*4+3
  int mloc = lane & 15;
  int base0 = (lane & 16) ? 8 : 0;
  long long e = (long long)blockIdx.x * 64 + mt * 16 + mloc;
  long long eg = e < N_EDGES ? e : (N_EDGES - 1);
  const bf16* As = hb + (size_t)row[eg] * 512;
  const bf16* Ad = hb + (size_t)col[eg] * 512;
  const size_t bstride = 8 * 512;  // (128/16)*32*16 elements per K-step
  const bf16* bpS = BpS + ((size_t)(nh * 4) * 32 + lane) * 16;
  const bf16* bpD = BpD + ((size_t)(nh * 4) * 32 + lane) * 16;
  v8f c[4];
#pragma unroll
  for (int t = 0; t < 4; t++)
#pragma unroll
    for (int v = 0; v < 8; v++) c[t][v] = 0.f;
#pragma unroll
  for (int ks = 0; ks < 16; ks++) {
    v8bf lo = *(const v8bf*)(As + ks * 32 + base0);
    v8bf hi = *(const v8bf*)(As + ks * 32 + base0 + 16);
    v16bf a;
#pragma unroll
    for (int i = 0; i < 8; i++) { a[i] = lo[i]; a[8 + i] = hi[i]; }
    const bf16* bk = bpS + (size_t)ks * bstride;
#pragma unroll
    for (int t = 0; t < 4; t++) {
      v16bf b = *(const v16bf*)(bk + (size_t)t * 512);
      c[t] = __builtin_amdgcn_wmma_f32_16x16x32_bf16(false, a, false, b, (short)0,
                                                     c[t], false, false);
    }
  }
#pragma unroll
  for (int ks = 0; ks < 16; ks++) {
    v8bf lo = *(const v8bf*)(Ad + ks * 32 + base0);
    v8bf hi = *(const v8bf*)(Ad + ks * 32 + base0 + 16);
    v16bf a;
#pragma unroll
    for (int i = 0; i < 8; i++) { a[i] = lo[i]; a[8 + i] = hi[i]; }
    const bf16* bk = bpD + (size_t)ks * bstride;
#pragma unroll
    for (int t = 0; t < 4; t++) {
      v16bf b = *(const v16bf*)(bk + (size_t)t * 512);
      c[t] = __builtin_amdgcn_wmma_f32_16x16x32_bf16(false, a, false, b, (short)0,
                                                     c[t], false, false);
    }
  }
  // rank-1 ea term + bias + relu -> LDS
  float eav[8];
#pragma unroll
  for (int v = 0; v < 8; v++) {
    long long ge = (long long)blockIdx.x * 64 + mt * 16 + base0 + v;
    eav[v] = ge < N_EDGES ? ea[ge] : 0.f;
  }
#pragma unroll
  for (int t = 0; t < 4; t++) {
    int n = (nh * 4 + t) * 16 + mloc;
    float w1e = W1last[n];
    float bb = b1[n];
#pragma unroll
    for (int v = 0; v < 8; v++) {
      int m = mt * 16 + base0 + v;
      hsm[m][n] = fmaxf(c[t][v] + eav[v] * w1e + bb, 0.f);
    }
  }
  __syncthreads();
  if (threadIdx.x < 64) {
    int m = threadIdx.x;
    long long ge = (long long)blockIdx.x * 64 + m;
    if (ge < N_EDGES) {
      float mu = 0.f;
      for (int j = 0; j < 128; j++) mu += hsm[m][j];
      mu *= (1.f / 128.f);
      float var = 0.f;
      for (int j = 0; j < 128; j++) { float t = hsm[m][j] - mu; var += t * t; }
      var *= (1.f / 128.f);
      float rs = rsqrtf(var + LN_EPS);
      float o = b2[0];
      for (int j = 0; j < 128; j++) o += ((hsm[m][j] - mu) * rs * g[j] + be[j]) * W2[j];
      out[ge] = o;
    }
  }
}

// ------------------------------------------------------------- final mean/std/abs
__global__ void k_reduce(const float* __restrict__ x, int n, float* __restrict__ acc, int mode) {
  __shared__ float sm[256];
  int i = blockIdx.x * 256 + threadIdx.x;
  float v = 0.f;
  if (i < n) {
    float t = x[i];
    if (mode == 0) v = t;
    else { float mu = acc[0] / n; float d = t - mu; v = d * d; }
  }
  sm[threadIdx.x] = v;
  __syncthreads();
  for (int s = 128; s > 0; s >>= 1) {
    if (threadIdx.x < s) sm[threadIdx.x] += sm[threadIdx.x + s];
    __syncthreads();
  }
  if (threadIdx.x == 0) atomicAdd(&acc[mode ? 1 : 0], sm[0]);
}
__global__ void k_final(const float* __restrict__ x, int n, const float* __restrict__ acc,
                        float* __restrict__ out) {
  int i = blockIdx.x * blockDim.x + threadIdx.x;
  if (i >= n) return;
  float mu = acc[0] / n;
  float rs = rsqrtf(acc[1] / (n - 1));
  out[i] = fabsf((x[i] - mu) * rs);
}

// ================================================================ host driver
static inline int cdiv(long long a, int b) { return (int)((a + b - 1) / b); }

extern "C" void kernel_launch(void* const* d_in, const int* in_sizes, int n_in,
                              void* d_out, int out_size, void* d_ws, size_t ws_size,
                              hipStream_t stream) {
  const float* x = (const float*)d_in[0];
  const long long* ei = (const long long*)d_in[1];
  const float* eattr = (const float*)d_in[2];
  const float* t1aW = (const float*)d_in[3];  const float* t1ab = (const float*)d_in[4];
  const float* t1bW = (const float*)d_in[5];  const float* t1bb = (const float*)d_in[6];
  const float* e1W1 = (const float*)d_in[7];  const float* e1b1 = (const float*)d_in[8];
  const float* e1g  = (const float*)d_in[9];  const float* e1be = (const float*)d_in[10];
  const float* e1W2 = (const float*)d_in[11]; const float* e1b2 = (const float*)d_in[12];
  const float* t2aW = (const float*)d_in[13]; const float* t2ab = (const float*)d_in[14];
  const float* t2bW = (const float*)d_in[15]; const float* t2bb = (const float*)d_in[16];
  const float* e2W1 = (const float*)d_in[17]; const float* e2b1 = (const float*)d_in[18];
  const float* e2g  = (const float*)d_in[19]; const float* e2be = (const float*)d_in[20];
  const float* e2W2 = (const float*)d_in[21]; const float* e2b2 = (const float*)d_in[22];
  const float* t3aW = (const float*)d_in[23]; const float* t3ab = (const float*)d_in[24];
  const float* t3bW = (const float*)d_in[25]; const float* t3bb = (const float*)d_in[26];
  const float* e3W1 = (const float*)d_in[27]; const float* e3b1 = (const float*)d_in[28];
  const float* e3g  = (const float*)d_in[29]; const float* e3be = (const float*)d_in[30];
  const float* e3W2 = (const float*)d_in[31]; const float* e3b2 = (const float*)d_in[32];

  const long long* row = ei;
  const long long* col = ei + N_EDGES;

  char* wsp = (char*)d_ws;
  size_t off = 0;
  auto carve = [&](size_t bytes) -> void* {
    void* p = wsp + off;
    off += (bytes + 255) & ~(size_t)255;
    return p;
  };
  float* featA = (float*)carve((size_t)N_NODES * 512 * 4);
  float* featB = (float*)carve((size_t)N_NODES * 512 * 4);
  float* hop   = (float*)carve((size_t)N_NODES * 512 * 4);
  bf16*  abf   = (bf16*)carve((size_t)N_NODES * 512 * 2);
  bf16*  wp    = (bf16*)carve((size_t)256 * 512 * 2);
  bf16*  wp2   = (bf16*)carve((size_t)512 * 128 * 2);
  float* deg   = (float*)carve((size_t)N_NODES * 4);
  float* nrm   = (float*)carve((size_t)N_EDGES * 4);
  float* eaA   = (float*)carve((size_t)N_EDGES * 4);
  float* eaB   = (float*)carve((size_t)N_EDGES * 4);
  float* acc   = (float*)carve(256);

  auto gcn_norm = [&](const float* w) {
    k_zero<<<cdiv(N_NODES, 256), 256, 0, stream>>>(deg, N_NODES);
    k_deg<<<cdiv(N_EDGES, 256), 256, 0, stream>>>(deg, col, w, N_EDGES);
    k_norm<<<cdiv(N_EDGES, 256), 256, 0, stream>>>(deg, row, col, w, nrm, N_EDGES);
  };

  auto scatter = [&](const float* src, float* dst, int C) {
    long long nz = (long long)N_NODES * C;
    k_zero<<<cdiv(nz, 256), 256, 0, stream>>>(dst, nz);
    if ((C & 3) == 0) {
      long long tot = (long long)N_EDGES * (C >> 2);
      k_scatter4<<<cdiv(tot, 256), 256, 0, stream>>>(src, dst, row, col, nrm, C, tot);
    } else {
      k_scatter1<<<cdiv(N_EDGES, 256), 256, 0, stream>>>(src, dst, row, col, nrm, N_EDGES);
    }
  };

  auto gemm_wmma = [&](const float* Af32, const float* W, const float* bias, float* Out,
                       int K, int Nc, int accumulate, int addbias) {
    long long na = (long long)N_NODES * K;
    k_f32_to_bf16<<<cdiv(na, 256), 256, 0, stream>>>(Af32, abf, na);
    int pt = (K / 32) * (Nc / 16) * 32;
    k_pack_w<<<cdiv(pt, 128), 128, 0, stream>>>(W, wp, K, Nc, Nc, 0, pt);
    dim3 g(N_NODES / 16, Nc / 64);
    k_wmma_gemm<<<g, 32, 0, stream>>>(abf, wp, bias, Out, K / 32, Nc, accumulate, addbias);
  };

  auto tag_wmma = [&](float* in, float* out, const float* W, const float* bias,
                      int Cin, int Cout) {
    gemm_wmma(in, W, bias, out, Cin, Cout, 0, 1);
    float* cur = in; float* nxt = hop;
    for (int k = 1; k <= 3; k++) {
      scatter(cur, nxt, Cin);
      gemm_wmma(nxt, W + (size_t)k * Cin * Cout, bias, out, Cin, Cout, 1, 0);
      float* t = cur; cur = nxt; nxt = t;
    }
  };

  auto tag_scalar = [&](float* in, float* out, const float* W, const float* bias,
                        int Cin, int Cout) {
    long long tot = (long long)N_NODES * Cout;
    k_sgemm<<<cdiv(tot, 256), 256, 0, stream>>>(in, W, bias, out, N_NODES, Cin, Cout, 0, 1);
    float* cur = in; float* nxt = hop;
    for (int k = 1; k <= 3; k++) {
      scatter(cur, nxt, Cin);
      k_sgemm<<<cdiv(tot, 256), 256, 0, stream>>>(nxt, W + (size_t)k * Cin * Cout, bias, out,
                                                  N_NODES, Cin, Cout, 1, 0);
      float* t = cur; cur = nxt; nxt = t;
    }
  };

  // ---------------- block 1 (scalar: tiny dims) ----------------
  k_copy<<<cdiv(N_NODES, 256), 256, 0, stream>>>(x, featB, N_NODES);
  gcn_norm(eattr);
  tag_scalar(featB, featA, t1aW, t1ab, 1, 8);
  k_relu<<<cdiv((long long)N_NODES * 8, 256), 256, 0, stream>>>(featA, (long long)N_NODES * 8);
  tag_scalar(featA, featB, t1bW, t1bb, 8, 32);
  k_edge_small<32, 8><<<cdiv(N_EDGES, 256), 256, 0, stream>>>(
      featB, row, col, eattr, e1W1, e1b1, e1g, e1be, e1W2, e1b2, eaA, N_EDGES);

  // ---------------- block 2 (WMMA node GEMMs) ----------------
  gcn_norm(eaA);
  tag_wmma(featB, featA, t2aW, t2ab, 32, 64);
  k_relu<<<cdiv((long long)N_NODES * 64, 256), 256, 0, stream>>>(featA, (long long)N_NODES * 64);
  tag_wmma(featA, featB, t2bW, t2bb, 64, 128);
  k_edge_small<128, 32><<<cdiv(N_EDGES, 256), 256, 0, stream>>>(
      featB, row, col, eaA, e2W1, e2b1, e2g, e2be, e2W2, e2b2, eaB, N_EDGES);

  // ---------------- block 3 (WMMA node GEMMs + fused WMMA edge MLP) ----------------
  gcn_norm(eaB);
  tag_wmma(featB, featA, t3aW, t3ab, 128, 256);
  k_relu<<<cdiv((long long)N_NODES * 256, 256), 256, 0, stream>>>(featA, (long long)N_NODES * 256);
  tag_wmma(featA, featB, t3bW, t3bb, 256, 512);

  long long n512 = (long long)N_NODES * 512;
  k_f32_to_bf16<<<cdiv(n512, 256), 256, 0, stream>>>(featB, abf, n512);
  int pt = (512 / 32) * (128 / 16) * 32;
  k_pack_w<<<cdiv(pt, 128), 128, 0, stream>>>(e3W1, wp, 512, 128, 128, 0, pt);
  k_pack_w<<<cdiv(pt, 128), 128, 0, stream>>>(e3W1, wp2, 512, 128, 128, 512, pt);
  k_edge_wmma<<<cdiv(N_EDGES, 64), 256, 0, stream>>>(
      abf, row, col, eaB, wp, wp2, e3W1 + (size_t)1024 * 128, e3b1, e3g, e3be, e3W2, e3b2, eaA);

  // ---------------- final (ea - mean)/std(ddof=1), abs ----------------
  k_zero<<<1, 256, 0, stream>>>(acc, 2);
  k_reduce<<<cdiv(N_EDGES, 256), 256, 0, stream>>>(eaA, N_EDGES, acc, 0);
  k_reduce<<<cdiv(N_EDGES, 256), 256, 0, stream>>>(eaA, N_EDGES, acc, 1);
  k_final<<<cdiv(N_EDGES, 256), 256, 0, stream>>>(eaA, N_EDGES, acc, (float*)d_out);
}
